// BiLSTMCRF_62491774157014
// MI455X (gfx1250) — matrix operand order repaired and checked by reading, not compiled, not run on hardware
//
#include <hip/hip_runtime.h>
#include <math.h>

// ---------------- problem constants (match reference) ----------------
#define B_      64
#define T_      512
#define WD_     256
#define H_      512
#define L_      50
#define START_  48
#define STOP_   49
#define MROWS   (B_ * T_)      // 32768 rows of the time-batch matrix
#define G4H     (4 * H_)       // 2048 gate width

// ---------------- WMMA types ----------------
typedef __bf16 bf16_t;
typedef bf16_t v8bf  __attribute__((ext_vector_type(8)));
typedef bf16_t v16bf __attribute__((ext_vector_type(16)));
typedef float  v8f   __attribute__((ext_vector_type(8)));

// TDM descriptor vector types (6-arg builtin form on this toolchain)
typedef unsigned int u32x4 __attribute__((ext_vector_type(4)));
typedef int          i32x4 __attribute__((ext_vector_type(4)));
typedef int          i32x8 __attribute__((ext_vector_type(8)));

#if defined(__has_builtin)
# if __has_builtin(__builtin_amdgcn_tensor_load_to_lds) && __has_builtin(__builtin_amdgcn_s_wait_tensorcnt)
#  define USE_TDM 1
# endif
#endif
#ifndef USE_TDM
# define USE_TDM 0
#endif

union ABfrag { v16bf v; v8bf h[2]; };

__device__ __forceinline__ v8f wmma_bf16(v16bf a, v16bf b, v8f c) {
  // D = A(16x32 bf16) * B(32x16 bf16) + C(16x16 f32)
  return __builtin_amdgcn_wmma_f32_16x16x32_bf16(
      /*neg_a=*/false, a, /*neg_b=*/false, b,
      /*c_mod=*/(short)0, c, /*reuse_a=*/false, /*reuse_b=*/false);
}

// A fragment (16x32, row-major source): lane m = lane&15, k-half = lane>>4.
// Lane holds K in [kc+8*kh, +8) and [kc+16+8*kh, +8)  (ISA 7.12.2).
__device__ __forceinline__ v16bf load_a16(const bf16_t* row, int kc, int lane) {
  int kh8 = ((lane >> 4) & 1) * 8;
  ABfrag u;
  u.h[0] = *(const v8bf*)(row + kc + kh8);
  u.h[1] = *(const v8bf*)(row + kc + 16 + kh8);
  return u.v;
}
// B fragment (32x16) for X @ W^T: column n = lane&15 of B == row n of W,
// lane holds K in [kc+8*kh, +8) and [kc+16+8*kh, +8), kh = lane>>4.
__device__ __forceinline__ v16bf load_b16(const bf16_t* wrow, int kc, int lane) {
  int kb = ((lane >> 4) & 1) * 8;
  ABfrag u;
  u.h[0] = *(const v8bf*)(wrow + kc + kb);
  u.h[1] = *(const v8bf*)(wrow + kc + kb + 16);
  return u.v;
}

__device__ __forceinline__ float selu_f(float x) {
  const float alpha = 1.6732632423543772f;
  const float scale = 1.0507009873554805f;
  return scale * (x > 0.f ? x : alpha * (__expf(x) - 1.f));
}
__device__ __forceinline__ float sigm(float x) { return 1.f / (1.f + __expf(-x)); }

// ---------------- weight conversion ----------------
__global__ void k_f32_bf16(const float* __restrict__ s, bf16_t* __restrict__ d, int n) {
  int i = blockIdx.x * blockDim.x + threadIdx.x;
  if (i < n) d[i] = (bf16_t)s[i];
}
// W_out (50x1024) -> zero-padded (64x1024) bf16
__global__ void k_wout_pad(const float* __restrict__ s, bf16_t* __restrict__ d) {
  int i = blockIdx.x * blockDim.x + threadIdx.x;
  if (i < 64 * 1024) {
    int r = i >> 10, col = i & 1023;
    d[i] = (r < L_) ? (bf16_t)s[r * 1024 + col] : (bf16_t)0.f;
  }
}

// ---------------- fused embedding gather + input GEMM + SELU ----------------
// X[m,n] = selu( sum_k concat(emb0[id0],emb1[id1])[k] * W_in[n,k] + b_in[n] )
__global__ void __launch_bounds__(256) k_embed_gemm(
    const int* __restrict__ xs, const float* __restrict__ emb0,
    const float* __restrict__ emb1, const bf16_t* __restrict__ Win_bf,
    const float* __restrict__ b_in, bf16_t* __restrict__ Xbf) {
  int mt = blockIdx.x;                       // 0..2047
  int wave = threadIdx.x >> 5, lane = threadIdx.x & 31;
  int m = mt * 16 + (lane & 15);
  const float* r0 = emb0 + (size_t)xs[m] * WD_;
  const float* r1 = emb1 + (size_t)xs[MROWS + m] * WD_;
  int kh8 = ((lane >> 4) & 1) * 8;
  v8f acc[4] = {};
  for (int kc = 0; kc < 2 * WD_; kc += 32) {
    int k0 = kc + kh8, k1 = kc + 16 + kh8;           // 8-elem runs never cross 256
    const float* p0 = (k0 < WD_) ? (r0 + k0) : (r1 + (k0 - WD_));
    const float* p1 = (k1 < WD_) ? (r0 + k1) : (r1 + (k1 - WD_));
    ABfrag a;
#pragma unroll
    for (int i = 0; i < 8; ++i) { a.h[0][i] = (bf16_t)p0[i]; a.h[1][i] = (bf16_t)p1[i]; }
#pragma unroll
    for (int nt = 0; nt < 4; ++nt) {
      int n = wave * 64 + nt * 16 + (lane & 15);
      v16bf b = load_b16(Win_bf + (size_t)n * (2 * WD_), kc, lane);
      acc[nt] = wmma_bf16(a.v, b, acc[nt]);
    }
  }
  int kh = (lane >> 4) & 1;
#pragma unroll
  for (int nt = 0; nt < 4; ++nt) {
    int n = wave * 64 + nt * 16 + (lane & 15);
    float bias = b_in[n];
#pragma unroll
    for (int i = 0; i < 8; ++i) {
      int r = mt * 16 + i + 8 * kh;                  // C/D layout: M = i + 8*kh
      Xbf[(size_t)r * H_ + n] = (bf16_t)selu_f(acc[nt][i] + bias);
    }
  }
}

// ---------------- batched input-to-hidden GEMMs (both directions) ----------------
// xp = X @ W_ih^T + bias.
// Block = 8 waves, block tile 32(m) x 512(n); A-tile staged in LDS (TDM when
// available), each wave computes 2 m-tiles x 4 n-tiles (8 WMMA per k-chunk).
#define PADA 520   // 512 bf16 + 8 bf16 (16B) row pad -> stride 1040B, bank-spread
__global__ void __launch_bounds__(256) k_gemm_xp(
    const bf16_t* __restrict__ Xbf, const bf16_t* __restrict__ Wf,
    const bf16_t* __restrict__ Wb, const float* __restrict__ bf_,
    const float* __restrict__ bb_, float* __restrict__ xpf,
    float* __restrict__ xpb) {
  __shared__ bf16_t sA[32 * PADA];
  int dir = blockIdx.z;
  const bf16_t* W = dir ? Wb : Wf;
  const float* bias = dir ? bb_ : bf_;
  float* xp = dir ? xpb : xpf;
  int mbase = blockIdx.x * 32;
  const bf16_t* Ag = Xbf + (size_t)mbase * H_;
  int tid = threadIdx.x, lane = tid & 31;

#if USE_TDM
  // Tensor Data Mover: 2D tile 32x512 of 2B elems, LDS padding 4 DWORDs per
  // 256 DWORDs (1024B row) -> padded row stride 1040B == PADA elements.
  if ((tid >> 5) == 0) {
    unsigned lds_off = (unsigned)(unsigned long long)(uintptr_t)(&sA[0]);
    unsigned long long ga = (unsigned long long)(uintptr_t)Ag;
    u32x4 g0;
    g0.x = (unsigned)__builtin_amdgcn_readfirstlane((int)1u);  // count=1
    g0.y = (unsigned)__builtin_amdgcn_readfirstlane((int)lds_off);
    g0.z = (unsigned)__builtin_amdgcn_readfirstlane((int)(unsigned)ga);
    g0.w = (unsigned)__builtin_amdgcn_readfirstlane(
        (int)(((unsigned)(ga >> 32) & 0x01FFFFFFu) | (2u << 30)));  // type=2
    i32x8 g1;
    g1[0] = __builtin_amdgcn_readfirstlane(
        (1 << 16) |        // data_size = 2 bytes
        (1 << 20) |        // pad_enable
        (7 << 22) |        // pad_interval: 256 DWORDs
        (3 << 25));        // pad_amount: 4 DWORDs
    g1[1] = __builtin_amdgcn_readfirstlane((int)(H_ << 16));   // tensor_dim0 lo16
    g1[2] = __builtin_amdgcn_readfirstlane((int)(32 << 16));   // tensor_dim1 lo16
    g1[3] = __builtin_amdgcn_readfirstlane((int)(H_ << 16));   // tile_dim0
    g1[4] = __builtin_amdgcn_readfirstlane((int)32);           // tile_dim1
    g1[5] = __builtin_amdgcn_readfirstlane((int)H_);           // dim0 stride lo32
    g1[6] = __builtin_amdgcn_readfirstlane(0);
    g1[7] = __builtin_amdgcn_readfirstlane(0);
    i32x4 z4 = {0, 0, 0, 0};
    i32x8 z8 = {0, 0, 0, 0, 0, 0, 0, 0};
    __builtin_amdgcn_tensor_load_to_lds(g0, g1, z4, z4, z8, 0);
    __builtin_amdgcn_s_wait_tensorcnt(0);
  }
#else
  // cooperative staged copy with identical padded layout
  for (int i = tid; i < 32 * 64; i += 256) {
    int r = i >> 6, cc = (i & 63) * 8;
    *(v8bf*)(sA + r * PADA + cc) = *(const v8bf*)(Ag + (size_t)r * H_ + cc);
  }
#endif
  __syncthreads();

  int wave = tid >> 5;
  int nbase = blockIdx.y * 512 + wave * 64;
  const bf16_t* sA0 = sA + (size_t)(lane & 15) * PADA;
  const bf16_t* sA1 = sA + (size_t)(16 + (lane & 15)) * PADA;
  v8f acc[2][4] = {};
  for (int kc = 0; kc < H_; kc += 32) {
    v16bf a0 = load_a16(sA0, kc, lane);
    v16bf a1 = load_a16(sA1, kc, lane);
#pragma unroll
    for (int nt = 0; nt < 4; ++nt) {
      int n = nbase + nt * 16 + (lane & 15);
      v16bf b = load_b16(W + (size_t)n * H_, kc, lane);
      acc[0][nt] = wmma_bf16(a0, b, acc[0][nt]);
      acc[1][nt] = wmma_bf16(a1, b, acc[1][nt]);
    }
  }
  int kh = (lane >> 4) & 1;
#pragma unroll
  for (int mi = 0; mi < 2; ++mi) {
#pragma unroll
    for (int nt = 0; nt < 4; ++nt) {
      int n = nbase + nt * 16 + (lane & 15);
      float bb = bias[n];
#pragma unroll
      for (int i = 0; i < 8; ++i) {
        int r = mbase + mi * 16 + i + 8 * kh;
        xp[(size_t)r * G4H + n] = acc[mi][nt][i] + bb;
      }
    }
  }
}

// ---------------- persistent bidirectional LSTM scan ----------------
#define NBD 16   // blocks per direction
__device__ __forceinline__ void dir_barrier(unsigned* bar, unsigned target, int tid) {
  __syncthreads();
  if (tid == 0) {
    __threadfence();
    atomicAdd(bar, 1u);
    while (*(volatile unsigned*)bar < target) __builtin_amdgcn_s_sleep(2);
    __threadfence();
  }
  __syncthreads();
}

__global__ void __launch_bounds__(256) k_lstm(
    const float* __restrict__ xpf, const float* __restrict__ xpb,
    const bf16_t* __restrict__ Whhf, const bf16_t* __restrict__ Whhb,
    const int* __restrict__ lens, bf16_t* __restrict__ hsf,
    bf16_t* __restrict__ hsb, bf16_t* hcur, float* cst, float* gbuf,
    unsigned* bars) {
  int dir = blockIdx.x / NBD;
  int blk = blockIdx.x % NBD;
  const float* xp = dir ? xpb : xpf;
  const bf16_t* W = dir ? Whhb : Whhf;
  bf16_t* hs = dir ? hsb : hsf;
  bf16_t* h = hcur + dir * (B_ * H_);
  float*  c = cst  + dir * (B_ * H_);
  float*  g = gbuf + dir * (B_ * G4H);
  unsigned* bar = bars + dir;
  unsigned epoch = 0;
  int tid = threadIdx.x, lane = tid & 31;
  int tdir = blk * 256 + tid;                         // 0..4095 within direction

  for (int e = tdir; e < B_ * H_; e += NBD * 256) { h[e] = (bf16_t)0.f; c[e] = 0.f; }
  dir_barrier(bar, epoch += NBD, tid);

  // 128 waves/direction; each wave: 2 m-tiles x 2 n-tiles (512 tiles total)
  int gw = blk * 8 + (tid >> 5);                      // 0..127
  int mt0 = (gw >> 6) * 2;                            // {0, 2}
  int np  = (gw & 63) * 2;                            // n-tile pair base
  const bf16_t* arow0 = h + (size_t)(mt0 * 16 + (lane & 15)) * H_;
  const bf16_t* arow1 = arow0 + 16 * H_;

  for (int s = 0; s < T_; ++s) {
    int t = dir ? (T_ - 1 - s) : s;
    // --- GEMM phase: g = h @ W_hh^T + xp[:,t,:] ---
    v8f acc[2][2] = {};
    for (int kc = 0; kc < H_; kc += 32) {
      v16bf a0 = load_a16(arow0, kc, lane);
      v16bf a1 = load_a16(arow1, kc, lane);
#pragma unroll
      for (int j = 0; j < 2; ++j) {
        int n = (np + j) * 16 + (lane & 15);
        v16bf b = load_b16(W + (size_t)n * H_, kc, lane);
        acc[0][j] = wmma_bf16(a0, b, acc[0][j]);
        acc[1][j] = wmma_bf16(a1, b, acc[1][j]);
      }
    }
    int kh = (lane >> 4) & 1;
#pragma unroll
    for (int mi = 0; mi < 2; ++mi) {
#pragma unroll
      for (int j = 0; j < 2; ++j) {
        int n = (np + j) * 16 + (lane & 15);
#pragma unroll
        for (int i = 0; i < 8; ++i) {
          int r = (mt0 + mi) * 16 + i + 8 * kh;       // batch index 0..63
          g[r * G4H + n] = acc[mi][j][i] + xp[((size_t)r * T_ + t) * G4H + n];
        }
      }
    }
    dir_barrier(bar, epoch += NBD, tid);
    // --- gate phase ---
    for (int e = tdir; e < B_ * H_; e += NBD * 256) {
      int bi = e >> 9, j = e & (H_ - 1);
      const float* gr = g + bi * G4H;
      float ig = gr[j], fg = gr[H_ + j], gg = gr[2 * H_ + j], og = gr[3 * H_ + j];
      float c_old = c[e];
      float c_new = sigm(fg) * c_old + sigm(ig) * tanhf(gg);
      float h_new = sigm(og) * tanhf(c_new);
      bool valid = t < lens[bi];
      float c2 = valid ? c_new : c_old;
      float h2 = valid ? h_new : (float)h[e];
      c[e] = c2;
      h[e] = (bf16_t)h2;
      hs[((size_t)bi * T_ + t) * H_ + j] = valid ? (bf16_t)h_new : (bf16_t)0.f;
    }
    dir_barrier(bar, epoch += NBD, tid);
  }
}

// ---------------- output projection: logits = selu(concat(hf,hb) @ Wout^T + b) ----
__global__ void __launch_bounds__(128) k_proj(
    const bf16_t* __restrict__ hsf, const bf16_t* __restrict__ hsb,
    const bf16_t* __restrict__ WoutP, const float* __restrict__ b_out,
    float* __restrict__ logits) {
  int mt = blockIdx.x;
  int wave = threadIdx.x >> 5, lane = threadIdx.x & 31;
  int m = mt * 16 + (lane & 15);
  const bf16_t* rf = hsf + (size_t)m * H_;
  const bf16_t* rb = hsb + (size_t)m * H_;
  int kh8 = ((lane >> 4) & 1) * 8;
  v8f acc = {};
  for (int kc = 0; kc < 2 * H_; kc += 32) {
    int k0 = kc + kh8, k1 = kc + 16 + kh8;
    const bf16_t* p0 = (k0 < H_) ? (rf + k0) : (rb + (k0 - H_));
    const bf16_t* p1 = (k1 < H_) ? (rf + k1) : (rb + (k1 - H_));
    ABfrag a;
    a.h[0] = *(const v8bf*)p0;
    a.h[1] = *(const v8bf*)p1;
    int n = wave * 16 + (lane & 15);
    acc = wmma_bf16(a.v, load_b16(WoutP + (size_t)n * (2 * H_), kc, lane), acc);
  }
  int n = wave * 16 + (lane & 15);
  if (n < L_) {
    float bias = b_out[n];
    int kh = (lane >> 4) & 1;
#pragma unroll
    for (int i = 0; i < 8; ++i) {
      int r = mt * 16 + i + 8 * kh;
      logits[(size_t)r * L_ + n] = selu_f(acc[i] + bias);
    }
  }
}

// ---------------- CRF forward algorithm (one block per batch element) ----------
__global__ void __launch_bounds__(64) k_crf_norm(
    const float* __restrict__ logits, const int* __restrict__ lens,
    const float* __restrict__ trans, float* __restrict__ norm) {
  __shared__ float tr[L_ * L_];
  __shared__ float alpha[L_];
  int b = blockIdx.x, tid = threadIdx.x;
  for (int i = tid; i < L_ * L_; i += 64) tr[i] = trans[i];
  if (tid < L_) alpha[tid] = (tid == START_) ? 0.f : -10000.f;
  __syncthreads();
  int len = lens[b];
  for (int t = 0; t < T_; ++t) {
    float a_new = 0.f;
    if (tid < L_) {
      float mx = -1e30f;
      for (int k = 0; k < L_; ++k) mx = fmaxf(mx, tr[tid * L_ + k] + alpha[k]);
      float s = 0.f;
      for (int k = 0; k < L_; ++k) s += __expf(tr[tid * L_ + k] + alpha[k] - mx);
      a_new = mx + __logf(s) + logits[((size_t)b * T_ + t) * L_ + tid];
    }
    __syncthreads();
    if (tid < L_ && t < len) alpha[tid] = a_new;
    __syncthreads();
  }
  if (tid == 0) {
    float mx = -1e30f;
    for (int j = 0; j < L_; ++j) mx = fmaxf(mx, alpha[j] + tr[STOP_ * L_ + j]);
    float s = 0.f;
    for (int j = 0; j < L_; ++j) s += __expf(alpha[j] + tr[STOP_ * L_ + j] - mx);
    norm[b] = mx + __logf(s);
  }
}

// ---------------- gold-path score + final loglik ----------------
__global__ void __launch_bounds__(64) k_crf_score(
    const int* __restrict__ y, const int* __restrict__ lens,
    const float* __restrict__ trans, const float* __restrict__ logits,
    const float* __restrict__ norm, float* __restrict__ out) {
  __shared__ float red[64];
  int b = blockIdx.x, tid = threadIdx.x;
  int len = lens[b];
  float s = 0.f;
  for (int i = tid; i <= len; i += 64) {
    int prev = (i == 0)   ? START_ : y[b * T_ + i - 1];
    int nxt  = (i == len) ? STOP_  : y[b * T_ + i];
    s += trans[nxt * L_ + prev];
  }
  for (int t = tid; t < len; t += 64)
    s += logits[((size_t)b * T_ + t) * L_ + y[b * T_ + t]];
  red[tid] = s;
  __syncthreads();
  for (int off = 32; off > 0; off >>= 1) {
    if (tid < off) red[tid] += red[tid + off];
    __syncthreads();
  }
  if (tid == 0) out[b] = red[0] - norm[b];
}

// ---------------- launcher ----------------
extern "C" void kernel_launch(void* const* d_in, const int* in_sizes, int n_in,
                              void* d_out, int out_size, void* d_ws, size_t ws_size,
                              hipStream_t stream) {
  (void)in_sizes; (void)n_in; (void)out_size; (void)ws_size;
  const int*   xs     = (const int*)d_in[0];
  const int*   y      = (const int*)d_in[1];
  const int*   lens   = (const int*)d_in[2];
  const float* emb0   = (const float*)d_in[3];
  const float* emb1   = (const float*)d_in[4];
  const float* W_in   = (const float*)d_in[5];
  const float* b_in   = (const float*)d_in[6];
  const float* W_ih_f = (const float*)d_in[7];
  const float* W_hh_f = (const float*)d_in[8];
  const float* b_f    = (const float*)d_in[9];
  const float* W_ih_b = (const float*)d_in[10];
  const float* W_hh_b = (const float*)d_in[11];
  const float* b_b    = (const float*)d_in[12];
  const float* W_out  = (const float*)d_in[13];
  const float* b_out  = (const float*)d_in[14];
  const float* trans  = (const float*)d_in[15];

  float* out    = (float*)d_out;
  float* loglik = out;            // 64 floats
  float* logits = out + B_;       // 64*512*50 floats

  char* w = (char*)d_ws;
  size_t off = 0;
  auto take = [&](size_t bytes) -> void* {
    void* p = w + off;
    off = (off + bytes + 255) & ~(size_t)255;
    return p;
  };
  unsigned* bars    = (unsigned*)take(256);
  float*    norm    = (float*)take(B_ * sizeof(float));
  bf16_t*   Win_bf  = (bf16_t*)take((size_t)H_ * (2 * WD_) * 2);
  bf16_t*   Wihf_bf = (bf16_t*)take((size_t)G4H * H_ * 2);
  bf16_t*   Wihb_bf = (bf16_t*)take((size_t)G4H * H_ * 2);
  bf16_t*   Whhf_bf = (bf16_t*)take((size_t)G4H * H_ * 2);
  bf16_t*   Whhb_bf = (bf16_t*)take((size_t)G4H * H_ * 2);
  bf16_t*   Wout_bf = (bf16_t*)take((size_t)64 * (2 * H_) * 2);
  bf16_t*   Xbf     = (bf16_t*)take((size_t)MROWS * H_ * 2);
  float*    xpf     = (float*)take((size_t)MROWS * G4H * 4);
  float*    xpb     = (float*)take((size_t)MROWS * G4H * 4);
  bf16_t*   hsf     = (bf16_t*)take((size_t)MROWS * H_ * 2);
  bf16_t*   hsb     = (bf16_t*)take((size_t)MROWS * H_ * 2);
  bf16_t*   hcur    = (bf16_t*)take((size_t)2 * B_ * H_ * 2);
  float*    cst     = (float*)take((size_t)2 * B_ * H_ * 4);
  float*    gbuf    = (float*)take((size_t)2 * B_ * G4H * 4);

  (void)hipMemsetAsync(bars, 0, 256, stream);

  int nWin = H_ * 2 * WD_;        // 262144
  int nWih = G4H * H_;            // 1048576
  k_f32_bf16<<<(nWin + 255) / 256, 256, 0, stream>>>(W_in, Win_bf, nWin);
  k_f32_bf16<<<(nWih + 255) / 256, 256, 0, stream>>>(W_ih_f, Wihf_bf, nWih);
  k_f32_bf16<<<(nWih + 255) / 256, 256, 0, stream>>>(W_ih_b, Wihb_bf, nWih);
  k_f32_bf16<<<(nWih + 255) / 256, 256, 0, stream>>>(W_hh_f, Whhf_bf, nWih);
  k_f32_bf16<<<(nWih + 255) / 256, 256, 0, stream>>>(W_hh_b, Whhb_bf, nWih);
  k_wout_pad<<<(64 * 1024 + 255) / 256, 256, 0, stream>>>(W_out, Wout_bf);

  k_embed_gemm<<<MROWS / 16, 256, 0, stream>>>(xs, emb0, emb1, Win_bf, b_in, Xbf);
  k_gemm_xp<<<dim3(MROWS / 32, 4, 2), 256, 0, stream>>>(Xbf, Wihf_bf, Wihb_bf,
                                                        b_f, b_b, xpf, xpb);
  k_lstm<<<2 * NBD, 256, 0, stream>>>(xpf, xpb, Whhf_bf, Whhb_bf, lens,
                                      hsf, hsb, hcur, cst, gbuf, bars);
  k_proj<<<MROWS / 16, 128, 0, stream>>>(hsf, hsb, Wout_bf, b_out, logits);
  k_crf_norm<<<B_, 64, 0, stream>>>(logits, lens, trans, norm);
  k_crf_score<<<B_, 64, 0, stream>>>(y, lens, trans, logits, norm, loglik);
}
